// FourierNeuralOperatorNet_53523882443326
// MI455X (gfx1250) — matrix operand order repaired and checked by reading, not compiled, not run on hardware
//
#include <hip/hip_runtime.h>
#include <hip/hip_bf16.h>
#include <math.h>

typedef __attribute__((ext_vector_type(16))) _Float16 v16h;
typedef __attribute__((ext_vector_type(8)))  _Float16 v8h;
typedef __attribute__((ext_vector_type(8)))  float    v8f;

#define PI_F 3.14159265358979323846f

__device__ __forceinline__ float gelu_f(float x) {
    return 0.5f * x * (1.0f + erff(x * 0.70710678118654752f));
}

// ---------------------------------------------------------------------------
// Generic WMMA GEMM:  Y[M,N] = act( beta*Y + W[M,K] @ X[K,N] + bias )
// Block tile 64(M) x 64(N), 256 threads = 8 wave32s; each wave computes a
// 16x32 output (two 16x16 WMMA accumulators sharing one A fragment).
// A staged [m][k]; B staged transposed [n][k] so both fragments are two
// aligned 16-byte LDS reads per lane (ds_load_b128 path).
// ---------------------------------------------------------------------------
#define GTM 64
#define GTN 64
__global__ __launch_bounds__(256) void k_gemm(
        const float* __restrict__ W, const float* __restrict__ X,
        float* __restrict__ Y, const float* __restrict__ bias,
        int M, int N, int K, int ldW, int beta, int act)
{
    __shared__ __align__(16) _Float16 sA [GTM][32];   // [m][k]
    __shared__ __align__(16) _Float16 sBt[GTN][32];   // [n][k] (transposed)

    const int tid  = threadIdx.x;
    const int lane = tid & 31;
    const int wave = tid >> 5;
    const int m0 = blockIdx.y * GTM;
    const int n0 = blockIdx.x * GTN;
    const int wm = (wave & 3) * 16;    // wave's M sub-tile (4 tiles)
    const int wn = (wave >> 2) * 32;   // wave's N range (2 waves x 32 cols)

    v8f acc[2] = {{}, {}};

    for (int k0 = 0; k0 < K; k0 += 32) {
        const bool fullA = (m0 + GTM <= M) && (k0 + 32 <= K);
        const bool fullB = (n0 + GTN <= N) && (k0 + 32 <= K);

        if (fullA) {
            // 64x32 floats, float2 per thread (rows 8-byte aligned: ldW even)
#pragma unroll
            for (int it = 0; it < 4; ++it) {
                int slot = tid + it * 256;          // 0..1023
                int mm = slot >> 4;                 // 16 float2 slots per row
                int kk = (slot & 15) * 2;
                float2 v = *(const float2*)(W + (size_t)(m0 + mm) * ldW + (k0 + kk));
                sA[mm][kk]     = (_Float16)v.x;
                sA[mm][kk + 1] = (_Float16)v.y;
            }
        } else {
            for (int idx = tid; idx < GTM * 32; idx += 256) {
                int mm = idx >> 5, kk = idx & 31;
                int gm = m0 + mm, gk = k0 + kk;
                float v = (gm < M && gk < K) ? W[(size_t)gm * ldW + gk] : 0.0f;
                sA[mm][kk] = (_Float16)v;
            }
        }

        if (fullB) {
            // 32(k) x 64(n) floats, float2 per thread along n (N even)
#pragma unroll
            for (int it = 0; it < 4; ++it) {
                int slot = tid + it * 256;          // 0..1023
                int kk = slot >> 5;                 // 32 float2 slots per k-row
                int nn = (slot & 31) * 2;
                float2 v = *(const float2*)(X + (size_t)(k0 + kk) * N + (n0 + nn));
                sBt[nn][kk]     = (_Float16)v.x;
                sBt[nn + 1][kk] = (_Float16)v.y;
            }
        } else {
            for (int idx = tid; idx < 32 * GTN; idx += 256) {
                int kk = idx >> 6, nn = idx & 63;
                int gk = k0 + kk, gn = n0 + nn;
                float v = (gk < K && gn < N) ? X[(size_t)gk * N + gn] : 0.0f;
                sBt[nn][kk] = (_Float16)v;
            }
        }

        // prefetch next K-tile of X (gfx1250 global_prefetch path)
        if (k0 + 32 < K)
            __builtin_prefetch(X + (size_t)(k0 + 32) * N + n0, 0, 1);

        __syncthreads();

        // A fragment: two aligned 16B runs per lane
        const int am = lane & 15;
        const int kh = (lane & 16) >> 1;            // 0 or 8
        v8h a_lo = *(const v8h*)&sA[wm + am][kh];
        v8h a_hi = *(const v8h*)&sA[wm + am][16 + kh];
        v16h a = __builtin_shufflevector(a_lo, a_hi,
                 0,1,2,3,4,5,6,7,8,9,10,11,12,13,14,15);

        // B fragments: k = kb..kb+7 and 16+kb..16+kb+7 in sBt row
        const int bn = lane & 15;
        const int kb = (lane & 16) >> 1;            // 0 or 8
#pragma unroll
        for (int t = 0; t < 2; ++t) {
            const int ncol = wn + t * 16 + bn;
            v8h b_lo = *(const v8h*)&sBt[ncol][kb];
            v8h b_hi = *(const v8h*)&sBt[ncol][16 + kb];
            v16h b = __builtin_shufflevector(b_lo, b_hi,
                     0,1,2,3,4,5,6,7,8,9,10,11,12,13,14,15);
            acc[t] = __builtin_amdgcn_wmma_f32_16x16x32_f16(
                         false, a, false, b, (short)0, acc[t], false, false);
        }
        __syncthreads();
    }

    // epilogue: C/D f32 16x16 layout (VGPR r -> M = r + 8*(lane>=16))
#pragma unroll
    for (int t = 0; t < 2; ++t) {
        const int n = n0 + wn + t * 16 + (lane & 15);
#pragma unroll
        for (int r = 0; r < 8; ++r) {
            int m = m0 + wm + r + ((lane & 16) >> 1);
            if (m < M && n < N) {
                float v = acc[t][r];
                if (bias) v += bias[m];
                size_t off = (size_t)m * N + n;
                if (beta) v += Y[off];
                if (act)  v = gelu_f(v);
                Y[off] = v;
            }
        }
    }
}

// ---------------------------------------------------------------------------
// Instance norm over [C, Npix]: one block per channel.
// ---------------------------------------------------------------------------
__global__ void k_inorm(const float* __restrict__ in, float* __restrict__ out,
                        const float* __restrict__ g, const float* __restrict__ b,
                        int Npix)
{
    const int c = blockIdx.x;
    const float* p = in + (size_t)c * Npix;
    float s = 0.0f, s2 = 0.0f;
    for (int i = threadIdx.x; i < Npix; i += 256) {
        float v = p[i]; s += v; s2 += v * v;
    }
    __shared__ float rs[256], rq[256];
    rs[threadIdx.x] = s; rq[threadIdx.x] = s2;
    __syncthreads();
    for (int o = 128; o > 0; o >>= 1) {
        if (threadIdx.x < o) {
            rs[threadIdx.x] += rs[threadIdx.x + o];
            rq[threadIdx.x] += rq[threadIdx.x + o];
        }
        __syncthreads();
    }
    const float mean = rs[0] / (float)Npix;
    const float var  = rq[0] / (float)Npix - mean * mean;
    const float sc = rsqrtf(var + 1e-6f) * g[c];
    const float sh = b[c] - mean * sc;
    float* q = out + (size_t)c * Npix;
    for (int i = threadIdx.x; i < Npix; i += 256)
        q[i] = p[i] * sc + sh;
}

// ---------------------------------------------------------------------------
// Elementwise helpers.
// ---------------------------------------------------------------------------
__global__ void k_gelu(float* __restrict__ p, size_t n) {
    size_t i = (size_t)blockIdx.x * blockDim.x + threadIdx.x;
    if (i < n) p[i] = gelu_f(p[i]);
}
__global__ void k_add(float* __restrict__ dst, const float* __restrict__ src, size_t n) {
    size_t i = (size_t)blockIdx.x * blockDim.x + threadIdx.x;
    if (i < n) dst[i] += src[i];
}

// ---------------------------------------------------------------------------
// Truncated DFT along last dim: out[r][k] = scale * sum_w in[r][w] e^{-2pi i k w/Wlen}
// ---------------------------------------------------------------------------
__global__ void k_dft_w(const float* __restrict__ in, float* __restrict__ out,
                        int R, int Wlen, int KX, float scale)
{
    int idx = blockIdx.x * blockDim.x + threadIdx.x;
    if (idx >= R * KX) return;
    int r = idx / KX, k = idx % KX;
    const float* p = in + (size_t)r * Wlen;
    float re = 0.0f, im = 0.0f;
    const float step = -2.0f * PI_F * (float)k / (float)Wlen;
    for (int w = 0; w < Wlen; ++w) {
        float sn, cs; sincosf(step * (float)w, &sn, &cs);
        float v = p[w];
        re += v * cs; im += v * sn;
    }
    out[2 * idx]     = re * scale;
    out[2 * idx + 1] = im * scale;
}

// DFT along H with retained-row mapping: in [C,Hin,KX,2] -> out [C,22,KX,2]
__global__ void k_dft_h(const float* __restrict__ in, float* __restrict__ out,
                        int Cc, int Hin, int KX, float scale)
{
    int idx = blockIdx.x * blockDim.x + threadIdx.x;
    if (idx >= Cc * 22 * KX) return;
    int c = idx / (22 * KX);
    int rem = idx % (22 * KX);
    int ky = rem / KX, kx = rem % KX;
    int f = (ky < 11) ? ky : (Hin - 22 + ky);
    const float* p = in + ((size_t)c * Hin * KX + kx) * 2;
    float re = 0.0f, im = 0.0f;
    const float step = -2.0f * PI_F * (float)f / (float)Hin;
    for (int hh = 0; hh < Hin; ++hh) {
        float sn, cs; sincosf(step * (float)hh, &sn, &cs);
        float a = p[(size_t)hh * KX * 2], b = p[(size_t)hh * KX * 2 + 1];
        re += a * cs - b * sn;
        im += a * sn + b * cs;
    }
    out[2 * idx]     = re * scale;
    out[2 * idx + 1] = im * scale;
}

// Per-mode complex channel mix: Yk[o,mode] = sum_i Xk[i,mode] * Wc[i,o,mode]
__global__ void k_specmix(const float* __restrict__ Xk, const float* __restrict__ Wc,
                          float* __restrict__ Yk)
{
    const int mode = blockIdx.x;   // 0..505
    const int o = threadIdx.x;     // 0..127
    __shared__ float xr[128], xi[128];
    xr[o] = Xk[(o * 506 + mode) * 2];
    xi[o] = Xk[(o * 506 + mode) * 2 + 1];
    __syncthreads();
    float re = 0.0f, im = 0.0f;
    for (int i = 0; i < 128; ++i) {
        const float* wp = Wc + (((size_t)i * 128 + o) * 506 + mode) * 2;
        float wr = wp[0], wi = wp[1];
        re += xr[i] * wr - xi[i] * wi;
        im += xr[i] * wi + xi[i] * wr;
    }
    Yk[(o * 506 + mode) * 2]     = re;
    Yk[(o * 506 + mode) * 2 + 1] = im;
}

// Inverse DFT along H: Z[c,h,kx] = sum_ky Yk[c,ky,kx] e^{+2pi i f(ky) h / Ho}
__global__ void k_idft_h(const float* __restrict__ Yk, float* __restrict__ Z,
                         int Cc, int Ho, int KX)
{
    int idx = blockIdx.x * blockDim.x + threadIdx.x;
    if (idx >= Cc * Ho * KX) return;
    int c = idx / (Ho * KX);
    int rem = idx % (Ho * KX);
    int hh = rem / KX, kx = rem % KX;
    float re = 0.0f, im = 0.0f;
    for (int ky = 0; ky < 22; ++ky) {
        int f = (ky < 11) ? ky : (Ho - 22 + ky);
        float ang = 2.0f * PI_F * (float)f * (float)hh / (float)Ho;
        float sn, cs; sincosf(ang, &sn, &cs);
        const float* p = Yk + (((size_t)c * 22 + ky) * KX + kx) * 2;
        float a = p[0], b = p[1];
        re += a * cs - b * sn;
        im += a * sn + b * cs;
    }
    Z[2 * idx] = re; Z[2 * idx + 1] = im;
}

// Hermitian inverse DFT along W: y[c,h,w] = Z0.re + 2*sum_{kx>=1}(Zr cos - Zi sin)
__global__ void k_idft_w(const float* __restrict__ Z, float* __restrict__ y,
                         int Cc, int Ho, int Wo, int KX)
{
    size_t idx = (size_t)blockIdx.x * blockDim.x + threadIdx.x;
    size_t total = (size_t)Cc * Ho * Wo;
    if (idx >= total) return;
    int w = (int)(idx % Wo);
    size_t row = idx / Wo;   // c*Ho + h
    const float* p = Z + row * KX * 2;
    float acc = p[0];
    const float step = 2.0f * PI_F * (float)w / (float)Wo;
    for (int kx = 1; kx < KX; ++kx) {
        float sn, cs; sincosf(step * (float)kx, &sn, &cs);
        acc += 2.0f * (p[2 * kx] * cs - p[2 * kx + 1] * sn);
    }
    y[idx] = acc;
}

// ---------------------------------------------------------------------------
// Host-side launcher
// ---------------------------------------------------------------------------
static inline void gemm(const float* W, const float* X, float* Y, const float* bias,
                        int M, int N, int K, int ldW, int beta, int act,
                        hipStream_t s)
{
    dim3 grid((N + GTN - 1) / GTN, (M + GTM - 1) / GTM);
    k_gemm<<<grid, 256, 0, s>>>(W, X, Y, bias, M, N, K, ldW, beta, act);
}

extern "C" void kernel_launch(void* const* d_in, const int* in_sizes, int n_in,
                              void* d_out, int out_size, void* d_ws, size_t ws_size,
                              hipStream_t stream)
{
    (void)in_sizes; (void)n_in; (void)out_size; (void)ws_size;

    const float* x        = (const float*)d_in[0];
    const float* pos      = (const float*)d_in[1];
    const float* enc_w    = (const float*)d_in[2];
    const float* enc_b    = (const float*)d_in[3];
    const float* n1g      = (const float*)d_in[4];
    const float* n1b      = (const float*)d_in[5];
    const float* n2g      = (const float*)d_in[6];
    const float* n2b      = (const float*)d_in[7];
    const float* spec_w   = (const float*)d_in[8];
    const float* inner_w  = (const float*)d_in[9];
    const float* inner_b  = (const float*)d_in[10];
    const float* icat_w   = (const float*)d_in[11];
    const float* ocat_w   = (const float*)d_in[12];
    const float* mlp_w1   = (const float*)d_in[13];
    const float* mlp_b1   = (const float*)d_in[14];
    const float* mlp_w2   = (const float*)d_in[15];
    const float* mlp_b2   = (const float*)d_in[16];
    const float* head0_w  = (const float*)d_in[17];
    const float* head0_b  = (const float*)d_in[18];
    const float* head1_w  = (const float*)d_in[19];
    const float* head1_b  = (const float*)d_in[20];
    float* out = (float*)d_out;

    const int C = 128, HH = 360, WW = 720, hh = 22, ww = 45, KX = 23;
    const size_t HW = (size_t)HH * WW;     // 259200
    const int LW = hh * ww;                // 990
    const int NM = 22 * KX;                // 506
    const size_t spec_stride = (size_t)C * C * NM * 2;

    // carve workspace (floats)
    float* ws    = (float*)d_ws;
    float* zfull = ws; ws += (size_t)C * HW;          // 33.18M
    float* tfull = ws; ws += (size_t)256 * HW;        // 66.36M
    float* Xwf   = ws; ws += (size_t)C * HH * KX * 2; // row-DFT (full) / Zrow (full)
    float* Xk    = ws; ws += (size_t)C * NM * 2;
    float* Yk    = ws; ws += (size_t)C * NM * 2;
    float* Zl    = ws; ws += (size_t)C * NM * 2;      // low-res row-DFT / Zrow
    float* zlow  = ws; ws += (size_t)C * LW;
    float* znl   = ws; ws += (size_t)C * LW;
    float* resl  = ws; ws += (size_t)C * LW;
    float* innl  = ws; ws += (size_t)C * LW;
    float* z2    = ws; ws += (size_t)C * LW;
    float* hidl  = ws; ws += (size_t)512 * LW;

    auto eltGrid = [](size_t n) { return (unsigned)((n + 255) / 256); };

    // ---- encoder + pos embed (full res, WMMA) ----
    gemm(enc_w, x, zfull, enc_b, C, (int)HW, 26, 26, 0, 0, stream);
    k_add<<<eltGrid((size_t)C * HW), 256, 0, stream>>>(zfull, pos, (size_t)C * HW);

    for (int i = 0; i < 8; ++i) {
        const float* g1 = n1g + i * C; const float* b1 = n1b + i * C;
        const float* g2 = n2g + i * C; const float* b2 = n2b + i * C;

        if (i == 0) {
            // full-res input path
            k_inorm<<<C, 256, 0, stream>>>(zfull, zfull, g1, b1, (int)HW);
            int R = C * HH;
            k_dft_w<<<eltGrid((size_t)R * KX), 256, 0, stream>>>(
                zfull, Xwf, R, WW, KX, 1.0f / (float)WW);
            k_dft_h<<<eltGrid((size_t)C * 22 * KX), 256, 0, stream>>>(
                Xwf, Xk, C, HH, KX, 1.0f / (float)HH);
        } else {
            if (i >= 1 && i <= 6)
                hipMemcpyAsync(resl, zlow, (size_t)C * LW * sizeof(float),
                               hipMemcpyDeviceToDevice, stream);
            float* nrm = (i == 7) ? zlow : znl;
            k_inorm<<<C, 256, 0, stream>>>(zlow, nrm, g1, b1, LW);
            int R = C * hh;
            k_dft_w<<<eltGrid((size_t)R * KX), 256, 0, stream>>>(
                nrm, Zl, R, ww, KX, 1.0f / (float)ww);
            k_dft_h<<<eltGrid((size_t)C * 22 * KX), 256, 0, stream>>>(
                Zl, Xk, C, hh, KX, 1.0f / (float)hh);
        }

        k_specmix<<<NM, 128, 0, stream>>>(Xk, spec_w + (size_t)i * spec_stride, Yk);

        if (i < 7) {
            // inverse transform to 22 x 45
            k_idft_h<<<eltGrid((size_t)C * hh * KX), 256, 0, stream>>>(Yk, Zl, C, hh, KX);
            k_idft_w<<<eltGrid((size_t)C * LW), 256, 0, stream>>>(Zl, zlow, C, hh, ww, KX);

            float* zcur = zlow;
            if (i >= 1 && i <= 6) {
                int j = i - 1;
                gemm(inner_w + (size_t)j * C * C, resl, innl, inner_b + j * C,
                     C, LW, C, C, 0, 0, stream);
                const float* IW = icat_w + (size_t)j * C * 2 * C;
                gemm(IW,       zcur, z2, nullptr, C, LW, C, 2 * C, 0, 0, stream);
                gemm(IW + C,   innl, z2, nullptr, C, LW, C, 2 * C, 1, 0, stream);
                zcur = z2;
            }
            k_gelu<<<eltGrid((size_t)C * LW), 256, 0, stream>>>(zcur, (size_t)C * LW);
            k_inorm<<<C, 256, 0, stream>>>(zcur, zcur, g2, b2, LW);

            // MLP (WMMA): 128 -> 512 -> 128
            gemm(mlp_w1 + (size_t)i * 512 * C, zcur, hidl, mlp_b1 + i * 512,
                 512, LW, C, C, 0, 1, stream);
            float* mout = (i >= 1 && i <= 6) ? znl : zlow;
            gemm(mlp_w2 + (size_t)i * C * 512, hidl, mout, mlp_b2 + i * C,
                 C, LW, 512, 512, 0, 0, stream);

            if (i >= 1 && i <= 6) {
                int j = i - 1;
                const float* OW = ocat_w + (size_t)j * C * 2 * C;
                gemm(OW,     mout, zlow, nullptr, C, LW, C, 2 * C, 0, 0, stream);
                gemm(OW + C, resl, zlow, nullptr, C, LW, C, 2 * C, 1, 0, stream);
            }
            // else: mout == zlow already holds layer output
        } else {
            // layer 7: inverse transform to full res
            k_idft_h<<<eltGrid((size_t)C * HH * KX), 256, 0, stream>>>(Yk, Xwf, C, HH, KX);
            k_idft_w<<<eltGrid((size_t)C * HW), 256, 0, stream>>>(Xwf, zfull, C, HH, WW, KX);
            k_gelu<<<eltGrid((size_t)C * HW), 256, 0, stream>>>(zfull, (size_t)C * HW);
            k_inorm<<<C, 256, 0, stream>>>(zfull, zfull, g2, b2, (int)HW);
        }
    }

    // ---- head: [z ; x] -> 256 -> gelu -> 26 (all WMMA) ----
    gemm(head0_w,       zfull, tfull, head0_b, 256, (int)HW, C,  C + 26, 0, 0, stream);
    gemm(head0_w + C,   x,     tfull, nullptr, 256, (int)HW, 26, C + 26, 1, 0, stream);
    k_gelu<<<eltGrid((size_t)256 * HW), 256, 0, stream>>>(tfull, (size_t)256 * HW);
    gemm(head1_w, tfull, out, head1_b, 26, (int)HW, 256, 256, 0, 0, stream);
}
